// Head_68058051772618
// MI455X (gfx1250) — matrix operand order repaired and checked by reading, not compiled, not run on hardware
//
#include <hip/hip_runtime.h>
#include <hip/hip_bf16.h>

// MI455X (gfx1250) causal attention head.
// Pipeline (intermediates in d_ws ~20.9 MB, resident in 192MB L2):
//   1) q = x@Wq, k = x@Wk, v = x@Wv   (f32 in -> f16 out, v_wmma_f32_16x16x32_f16)
//   2) flash attention (causal, online softmax) q,k,v f16 -> O f32
//      K tiles staged via GLOBAL_LOAD_ASYNC_TO_LDS_B128, double-buffered
//   3) out = O @ Wp                   (f32 out to d_out)

typedef _Float16 v16h __attribute__((ext_vector_type(16)));
typedef float    v8f  __attribute__((ext_vector_type(8)));

#define WMMA16x16x32(a, b, c) \
  __builtin_amdgcn_wmma_f32_16x16x32_f16(false, (a), false, (b), (short)0, (c), false, false)

static constexpr int Tt = 4096;
static constexpr int Dd = 128;

// Async copy of 16B global -> LDS (tracked by ASYNCcnt, not LOADcnt).
__device__ __forceinline__ void async_copy_b128(unsigned lds_off, const void* gptr) {
  asm volatile("global_load_async_to_lds_b128 %0, %1, off"
               :: "v"(lds_off), "v"(gptr) : "memory");
}

// ---------------------------------------------------------------------------
// GEMM: C[M x 128] = X[M x KD] (f32) @ W[KD x 128] (f32), M-tile 128 per block.
// 8 waves; wave w owns rows 16w..16w+15 and all 8 N-tiles (8 v8f accumulators).
// A fragment: direct from global f32, converted to f16 (layout: lanes 0-15
// carry K 0..15; lanes 16-31 carry K 16..31). B: W staged transposed in LDS
// so each fragment is two contiguous 16B ds loads.
// ---------------------------------------------------------------------------
template <int KD, bool OUT_HALF>
__global__ __launch_bounds__(256) void gemm_xw(const float* __restrict__ X,
                                               const float* __restrict__ W,
                                               void* __restrict__ outp) {
  __shared__ alignas(16) _Float16 lds_wT[128 * 32];  // [col][k] 8KB

  const int tid  = threadIdx.x;
  const int wave = tid >> 5, lane = tid & 31;
  const int half = lane >> 4, ln = lane & 15;
  const int m0   = blockIdx.x * 128;
  const int rowA = m0 + wave * 16 + ln;

  const v8f z = {0.f, 0.f, 0.f, 0.f, 0.f, 0.f, 0.f, 0.f};
  v8f acc[8];
#pragma unroll
  for (int nt = 0; nt < 8; ++nt) acc[nt] = z;

  for (int k0 = 0; k0 < KD; k0 += 32) {
    __syncthreads();
    // Stage W[k0..k0+31][0..127] transposed into LDS as f16 (coalesced reads).
#pragma unroll
    for (int i = 0; i < 16; ++i) {
      int e = tid + i * 256;       // 0..4095
      int kk = e >> 7, c = e & 127;
      lds_wT[c * 32 + kk] = (_Float16)W[(size_t)(k0 + kk) * 128 + c];
    }
    __syncthreads();

    // A fragment: 16 f32 from global (4x float4), convert to f16.
    alignas(16) float xv[16];
    const float* xp = X + (size_t)rowA * KD + k0 + half * 16;
    *(float4*)(xv + 0)  = ((const float4*)xp)[0];
    *(float4*)(xv + 4)  = ((const float4*)xp)[1];
    *(float4*)(xv + 8)  = ((const float4*)xp)[2];   // K offset +8
    *(float4*)(xv + 12) = ((const float4*)xp)[3];
    __builtin_prefetch(xp + 32, 0, 0);              // next k-step (L2-resident x)
    v16h a;
#pragma unroll
    for (int j = 0; j < 16; ++j) a[j] = (_Float16)xv[j];

#pragma unroll
    for (int nt = 0; nt < 8; ++nt) {
      v16h bf;
      const _Float16* wp = &lds_wT[(nt * 16 + ln) * 32 + half * 16];
      ((float4*)&bf)[0] = ((const float4*)wp)[0];
      ((float4*)&bf)[1] = ((const float4*)wp)[1];
      acc[nt] = WMMA16x16x32(a, bf, acc[nt]);
    }
  }

  // C layout: lanes 0-15 -> M=v, lanes 16-31 -> M=8+v; N = lane&15.
  const int mbase = m0 + wave * 16 + half * 8;
#pragma unroll
  for (int nt = 0; nt < 8; ++nt) {
#pragma unroll
    for (int v = 0; v < 8; ++v) {
      const size_t idx = (size_t)(mbase + v) * 128 + nt * 16 + ln;
      if (OUT_HALF) ((_Float16*)outp)[idx] = (_Float16)acc[nt][v];
      else          ((float*)outp)[idx]    = acc[nt][v];
    }
  }
}

// ---------------------------------------------------------------------------
// Flash attention, causal. One block = 128 query rows of one batch; 8 waves,
// wave w owns q rows 16w..16w+15. Key tiles of 64, K staged with async
// global->LDS copies (double-buffered); V staged through VGPRs (transpose).
// All waves execute the same iteration count so __syncthreads is uniform.
// ---------------------------------------------------------------------------
__global__ __launch_bounds__(256) void flash_attn(const _Float16* __restrict__ Q,
                                                  const _Float16* __restrict__ K,
                                                  const _Float16* __restrict__ V,
                                                  float* __restrict__ O) {
  __shared__ alignas(16) _Float16 lds_k[2][64 * 128];  // [s][d] 2x16KB double-buffered
  __shared__ alignas(16) _Float16 lds_vT[128 * 64];    // [d][s] 16KB
  __shared__ alignas(16) _Float16 lds_p[8][16 * 64];   // per-wave P 16KB

  const int tid  = threadIdx.x;
  const int wave = tid >> 5, lane = tid & 31;
  const int half = lane >> 4, ln = lane & 15;
  const int blk  = blockIdx.x;
  const int b = blk >> 5, qt = blk & 31;   // 32 query tiles per batch
  const int q0 = qt * 128;

  const _Float16* Qb = Q + (size_t)b * Tt * Dd;
  const _Float16* Kb = K + (size_t)b * Tt * Dd;
  const _Float16* Vb = V + (size_t)b * Tt * Dd;
  float*          Ob = O + (size_t)b * Tt * Dd;
  const int qrow_w = q0 + wave * 16;

  // Issue the 4 async 16B copies of this thread's share of one K tile.
  // LDS layout is row-major [s][d]: byte offset = 16*chunk exactly.
  auto issue_k_tile = [&](int tile, int buf) {
    const unsigned kbase = (unsigned)(size_t)(const void*)&lds_k[buf][0];
#pragma unroll
    for (int i = 0; i < 4; ++i) {
      int chunk = tid + i * 256;  // 1024 chunks of 8 f16
      const void* g = &Kb[(size_t)(tile * 64 + (chunk >> 4)) * Dd + (chunk & 15) * 8];
      async_copy_b128(kbase + chunk * 16, g);
    }
  };

  // Preload Q fragments for the 4 k-steps over d (loop invariant).
  v16h qf[4];
#pragma unroll
  for (int ks = 0; ks < 4; ++ks) {
    const _Float16* qp = Qb + (size_t)(qrow_w + ln) * Dd + ks * 32 + half * 16;
    ((float4*)&qf[ks])[0] = *(const float4*)qp;
    ((float4*)&qf[ks])[1] = *(const float4*)(qp + 8);
  }

  const v8f z = {0.f, 0.f, 0.f, 0.f, 0.f, 0.f, 0.f, 0.f};
  v8f o[8];
#pragma unroll
  for (int dt = 0; dt < 8; ++dt) o[dt] = z;
  v8f mrow, lrow;
#pragma unroll
  for (int v = 0; v < 8; ++v) { mrow[v] = -1e30f; lrow[v] = 0.f; }

  const float scale = 0.08838834764831845f;  // 1/sqrt(128)
  const int nkt = (q0 + 128) >> 6;           // key tiles up to causal bound (>=2)

  issue_k_tile(0, 0);  // prologue: K tile 0 in flight

  for (int kt = 0; kt < nkt; ++kt) {
    const int s0 = kt * 64;
    __syncthreads();  // previous iteration's LDS reads done

    // Stage V tile transposed through VGPRs (async copies can't transpose).
#pragma unroll
    for (int i = 0; i < 4; ++i) {
      int chunk = tid + i * 256;
      int s = chunk >> 4, c8 = (chunk & 15) * 8;
      float4 t = *(const float4*)&Vb[(size_t)(s0 + s) * Dd + c8];
      alignas(16) _Float16 tv[8];
      *(float4*)tv = t;
#pragma unroll
      for (int j = 0; j < 8; ++j) lds_vT[(c8 + j) * 64 + s] = tv[j];
    }

    // Prefetch next K tile into the other buffer (clamped at the end so the
    // asynccnt wait immediate stays a constant 4 in-flight newer copies).
    const int nxt = (kt + 1 < nkt) ? kt + 1 : nkt - 1;
    issue_k_tile(nxt, (kt + 1) & 1);
    // Async loads complete in order: <=4 outstanding means tile kt landed.
    asm volatile("s_wait_asynccnt 0x4" ::: "memory");
    __syncthreads();

    const _Float16* kb = &lds_k[kt & 1][0];

    // S = Q K^T  (16 x 64 per wave): 4 N-tiles x 4 k-steps.
    v8f sacc[4];
#pragma unroll
    for (int nt = 0; nt < 4; ++nt) sacc[nt] = z;
#pragma unroll
    for (int nt = 0; nt < 4; ++nt) {
#pragma unroll
      for (int ks = 0; ks < 4; ++ks) {
        v16h bf;
        const _Float16* kp = &kb[(nt * 16 + ln) * 128 + ks * 32 + half * 16];
        ((float4*)&bf)[0] = ((const float4*)kp)[0];
        ((float4*)&bf)[1] = ((const float4*)kp)[1];
        sacc[nt] = WMMA16x16x32(qf[ks], bf, sacc[nt]);
      }
    }

    // Scale + causal mask + running row max (rows live on fixed elements
    // within each 16-lane half, so xor-1/2/4/8 butterflies stay in-half).
    v8f mnew = mrow;
#pragma unroll
    for (int nt = 0; nt < 4; ++nt) {
#pragma unroll
      for (int v = 0; v < 8; ++v) {
        float val = sacc[nt][v] * scale;
        const int col = s0 + nt * 16 + ln;
        const int row = qrow_w + half * 8 + v;
        val = (col <= row) ? val : -1e30f;
        sacc[nt][v] = val;
        mnew[v] = fmaxf(mnew[v], val);
      }
    }
#pragma unroll
    for (int msk = 1; msk <= 8; msk <<= 1)
#pragma unroll
      for (int v = 0; v < 8; ++v)
        mnew[v] = fmaxf(mnew[v], __shfl_xor(mnew[v], msk, 32));

    // P = exp(S - mnew), row sums, rescale running state.
    v8f psum = z;
#pragma unroll
    for (int nt = 0; nt < 4; ++nt)
#pragma unroll
      for (int v = 0; v < 8; ++v) {
        float p = __expf(sacc[nt][v] - mnew[v]);
        sacc[nt][v] = p;
        psum[v] += p;
      }
#pragma unroll
    for (int msk = 1; msk <= 8; msk <<= 1)
#pragma unroll
      for (int v = 0; v < 8; ++v) psum[v] += __shfl_xor(psum[v], msk, 32);
#pragma unroll
    for (int v = 0; v < 8; ++v) {
      const float corr = __expf(mrow[v] - mnew[v]);
      lrow[v] = lrow[v] * corr + psum[v];
      mrow[v] = mnew[v];
#pragma unroll
      for (int dt = 0; dt < 8; ++dt) o[dt][v] *= corr;
    }

    // C-layout -> A-layout for P via per-wave LDS region.
#pragma unroll
    for (int nt = 0; nt < 4; ++nt)
#pragma unroll
      for (int v = 0; v < 8; ++v)
        lds_p[wave][(half * 8 + v) * 64 + nt * 16 + ln] = (_Float16)sacc[nt][v];
    __syncthreads();

    // O += P V : 2 k-steps over s, 8 d-tiles.
#pragma unroll
    for (int ks2 = 0; ks2 < 2; ++ks2) {
      v16h pf;
      const _Float16* pp = &lds_p[wave][ln * 64 + ks2 * 32 + half * 16];
      ((float4*)&pf)[0] = *(const float4*)pp;
      ((float4*)&pf)[1] = *(const float4*)(pp + 8);
#pragma unroll
      for (int dt = 0; dt < 8; ++dt) {
        v16h vf;
        const _Float16* vp = &lds_vT[(dt * 16 + ln) * 64 + ks2 * 32 + half * 16];
        ((float4*)&vf)[0] = ((const float4*)vp)[0];
        ((float4*)&vf)[1] = ((const float4*)vp)[1];
        o[dt] = WMMA16x16x32(pf, vf, o[dt]);
      }
    }
  }

  // Drain the final (redundant) prefetch before wave teardown.
  asm volatile("s_wait_asynccnt 0x0" ::: "memory");

  // Normalize and store f32.
#pragma unroll
  for (int v = 0; v < 8; ++v) {
    const float inv = 1.0f / lrow[v];
    const size_t rbase = (size_t)(qrow_w + half * 8 + v) * Dd;
#pragma unroll
    for (int dt = 0; dt < 8; ++dt) Ob[rbase + dt * 16 + ln] = o[dt][v] * inv;
  }
}

// ---------------------------------------------------------------------------
extern "C" void kernel_launch(void* const* d_in, const int* in_sizes, int n_in,
                              void* d_out, int out_size, void* d_ws, size_t ws_size,
                              hipStream_t stream) {
  (void)in_sizes; (void)n_in; (void)out_size; (void)ws_size;
  const float* x  = (const float*)d_in[0];
  const float* Wk = (const float*)d_in[1];
  const float* Wq = (const float*)d_in[2];
  const float* Wv = (const float*)d_in[3];
  const float* Wp = (const float*)d_in[4];
  float* out = (float*)d_out;

  const int M = 4 * Tt;  // 16384 rows
  _Float16* qh = (_Float16*)d_ws;
  _Float16* kh = qh + (size_t)M * Dd;
  _Float16* vh = kh + (size_t)M * Dd;
  float* of32  = (float*)(vh + (size_t)M * Dd);  // 16B-aligned (12MB offset)

  dim3 block(256);
  dim3 grid(M / 128);

  // QKV projections (x re-reads hit L2: 64MB < 192MB).
  gemm_xw<1024, true><<<grid, block, 0, stream>>>(x, Wq, (void*)qh);
  gemm_xw<1024, true><<<grid, block, 0, stream>>>(x, Wk, (void*)kh);
  gemm_xw<1024, true><<<grid, block, 0, stream>>>(x, Wv, (void*)vh);

  // Causal flash attention (one block per 128 query rows per batch).
  flash_attn<<<grid, block, 0, stream>>>(qh, kh, vh, of32);

  // Output projection to f32.
  gemm_xw<128, false><<<grid, block, 0, stream>>>(of32, Wp, (void*)out);
}
